// ESMDecoderHalfMOELayer_9483287790086
// MI455X (gfx1250) — matrix operand order repaired and checked by reading, not compiled, not run on hardware
//
#include <hip/hip_runtime.h>
#include <hip/hip_bf16.h>

// ---------------------------------------------------------------------------
// MI455X (gfx1250): bf16 WMMA + Tensor Data Mover (TDM) double-buffered GEMMs,
// WMMA flash attention, fused LN / gating. wave32 throughout.
// ---------------------------------------------------------------------------

#define BB 8
#define SS 2048
#define EE 512
#define HH_ 8
#define DD 64
#define NEXPE 16
#define BS_ (BB * SS)          // 16384 token rows

typedef __bf16 bf16;
typedef __attribute__((ext_vector_type(16))) __bf16 bf16v16;
typedef __attribute__((ext_vector_type(8)))  float  v8f;
typedef __attribute__((ext_vector_type(4)))  unsigned int u32x4;
typedef __attribute__((ext_vector_type(2)))  unsigned int u32x2;
typedef __attribute__((ext_vector_type(8)))  unsigned int u32x8;

union Frag { bf16v16 v; u32x4 u[2]; };

__device__ __forceinline__ v8f wmma_bf16(const Frag& a, const Frag& b, v8f c) {
  return __builtin_amdgcn_wmma_f32_16x16x32_bf16(false, a.v, false, b.v,
                                                 (short)0, c, false, false);
}

// ---------------------------------------------------------------------------
// TDM: issue a 2-D bf16 tile load (tile_dim0 = 32 elems along K, `rows` lines,
// line stride 512 elems) into LDS. Descriptor per CDNA5 ISA ch.8 (D# groups).
// Issued once per wave (EXEC ignored); completion via TENSORcnt.
// ---------------------------------------------------------------------------
__device__ __forceinline__ void tdm_load_2d(unsigned int lds_addr,
                                            const void* gptr,
                                            unsigned int tile_rows) {
  const unsigned long long ga = (unsigned long long)(uintptr_t)gptr;
  u32x4 g0;
  g0.x = 1u;                                    // count=1 (valid user D#)
  g0.y = lds_addr;                              // LDS byte address
  g0.z = (unsigned int)ga;                      // global_addr[31:0]
  g0.w = (unsigned int)((ga >> 32) & 0x01FFFFFFu) | (2u << 30); // [56:32]|type=2
  u32x8 g1;
  g1.s0 = 1u << 16;                             // data_size=1 (2 bytes), mask=0
  g1.s1 = (512u & 0xFFFFu) << 16;               // tensor_dim0 lo16 (=512)
  g1.s2 = (512u >> 16) | (16384u & 0xFFFFu) << 16; // dim0 hi | tensor_dim1 lo
  g1.s3 = (16384u >> 16) | (32u << 16);         // dim1 hi | tile_dim0 = 32
  g1.s4 = tile_rows & 0xFFFFu;                  // tile_dim1 (tile_dim2 = 0)
  g1.s5 = 512u;                                 // tensor_dim0_stride lo32
  g1.s6 = 0u;                                   // stride hi | dim1_stride lo
  g1.s7 = 0u;
  asm volatile("tensor_load_to_lds %0, %1" :: "s"(g0), "s"(g1) : "memory");
}

// ---------------------------------------------------------------------------
// fp32 -> bf16 conversion
// ---------------------------------------------------------------------------
__global__ __launch_bounds__(256) void cvt_bf16_kernel(const float* __restrict__ s,
                                                       bf16* __restrict__ d, int n) {
  int i = blockIdx.x * 256 + threadIdx.x;
  if (i < n) d[i] = (bf16)s[i];
}

// ---------------------------------------------------------------------------
// TDM double-buffered bf16 GEMM: OUT[M,N] = X[M,512] * W[N,512]^T + bias
//  block = 256 threads (8 waves, 2x4), block tile 64(M) x 128(N),
//  wave tile 32x32 = 4 x v_wmma_f32_16x16x32_bf16 per 32-deep K chunk.
//  Wave 0 DMA-prefetches the next chunk's A/B tiles with tensor_load_to_lds.
// MODE 0: QKV  (N=1536) -> split q(*1/sqrt(D))/k/v bf16
// MODE 1: dense -> fp32 out (+bias)
// MODE 2: MoE  -> loop NE experts, total += member * (gemm + bias_e), fp32 out
// ---------------------------------------------------------------------------
template <int MODE, int NE>
__global__ __launch_bounds__(256) void gemm_tdm_kernel(
    const bf16* __restrict__ X, const bf16* __restrict__ W,
    const float* __restrict__ bias, float* __restrict__ outF,
    bf16* __restrict__ oq, bf16* __restrict__ ok_, bf16* __restrict__ ov,
    const float* __restrict__ member) {
  __shared__ __align__(16) bf16 Abuf[2][64 * 32];    // 8 KB
  __shared__ __align__(16) bf16 Bbuf[2][128 * 32];   // 16 KB

  const int tid = threadIdx.x, lane = tid & 31, w = tid >> 5;
  const int wr = w >> 2, wc = w & 3;                 // 2 x 4 wave grid
  const int mbase = blockIdx.x * 64, nbase = blockIdx.y * 128;
  const int nl = lane & 15, hh = lane >> 4;
  const int m0 = wr * 32, n0 = wc * 32;

  const v8f vzero = {};
  v8f acc[2][2] = {};
  float total[2][2][8] = {};

  const int NSTEP = NE * 16;

  auto issue = [&](int step, int buf) {
    const int e = step >> 4, kc = step & 15;
    tdm_load_2d((unsigned int)(uintptr_t)&Abuf[buf][0],
                X + (size_t)mbase * 512 + kc * 32, 64u);
    tdm_load_2d((unsigned int)(uintptr_t)&Bbuf[buf][0],
                W + (size_t)e * (512 * 512) + (size_t)nbase * 512 + kc * 32,
                128u);
  };

  if (w == 0) issue(0, 0);

  for (int step = 0; step < NSTEP; ++step) {
    const int buf = step & 1;
    if (w == 0) {
      if (step + 1 < NSTEP) {
        issue(step + 1, buf ^ 1);                 // prefetch next chunk
        __builtin_amdgcn_s_wait_tensorcnt(2);     // current buf's pair landed
      } else {
        __builtin_amdgcn_s_wait_tensorcnt(0);
      }
    }
    __syncthreads();

    Frag a[2], b[2];
#pragma unroll
    for (int mi = 0; mi < 2; ++mi) {
      const bf16* ap = &Abuf[buf][(m0 + mi * 16 + nl) * 32];
      a[mi].u[0] = *(const u32x4*)(ap + 16 * hh);
      a[mi].u[1] = *(const u32x4*)(ap + 16 * hh + 8);
    }
#pragma unroll
    for (int ni = 0; ni < 2; ++ni) {
      const bf16* bp = &Bbuf[buf][(n0 + ni * 16 + nl) * 32 + 16 * hh];
      b[ni].u[0] = *(const u32x4*)bp;
      b[ni].u[1] = *(const u32x4*)(bp + 8);
    }
#pragma unroll
    for (int mi = 0; mi < 2; ++mi)
#pragma unroll
      for (int ni = 0; ni < 2; ++ni)
        acc[mi][ni] = wmma_bf16(a[mi], b[ni], acc[mi][ni]);
    __syncthreads();

    if ((step & 15) == 15) {  // end of one 512-deep K pass (one expert)
      const int e = step >> 4;
#pragma unroll
      for (int mi = 0; mi < 2; ++mi) {
#pragma unroll
        for (int ni = 0; ni < 2; ++ni) {
          const int gcol = nbase + n0 + ni * 16 + nl;
#pragma unroll
          for (int r = 0; r < 8; ++r) {
            const int grow = mbase + m0 + mi * 16 + r + 8 * hh;
            const size_t rowoff = (size_t)grow * 512;
            if (MODE == 0) {
              float val = acc[mi][ni][r] + bias[gcol];
              if (gcol < 512)       oq[rowoff + gcol]          = (bf16)(val * 0.125f);
              else if (gcol < 1024) ok_[rowoff + (gcol - 512)] = (bf16)val;
              else                  ov[rowoff + (gcol - 1024)] = (bf16)val;
            } else if (MODE == 1) {
              outF[rowoff + gcol] = acc[mi][ni][r] + bias[gcol];
            } else {
              total[mi][ni][r] += member[(size_t)grow * NEXPE + e] *
                                  (acc[mi][ni][r] + bias[e * 512 + gcol]);
            }
          }
          if (MODE == 2) acc[mi][ni] = vzero;
        }
      }
    }
  }

  if (MODE == 2) {
#pragma unroll
    for (int mi = 0; mi < 2; ++mi)
#pragma unroll
      for (int ni = 0; ni < 2; ++ni) {
        const int gcol = nbase + n0 + ni * 16 + nl;
#pragma unroll
        for (int r = 0; r < 8; ++r) {
          const int grow = mbase + m0 + mi * 16 + r + 8 * hh;
          outF[(size_t)grow * 512 + gcol] = total[mi][ni][r];
        }
      }
  }
}

// ---------------------------------------------------------------------------
// Flash attention: block = (64 q rows, head, batch), 4 waves / 128 threads.
// Each wave owns 16 q rows; 32-key chunks staged in LDS; online softmax.
// ctx may alias qin (each block reads only its own q slice before writing).
// ---------------------------------------------------------------------------
__global__ __launch_bounds__(128) void attn_fa_kernel(const bf16* qin,
                                                      const bf16* kin,
                                                      const bf16* vin,
                                                      bf16* ctx) {
  __shared__ __align__(16) bf16 Ks[32 * 64];       // [key][d]
  __shared__ __align__(16) bf16 Vst[64 * 32];      // [d][key]  (transposed)
  __shared__ __align__(16) bf16 Ps[4 * 16 * 32];   // per-wave P tile
  __shared__ float st_mc[64], st_mn[64], st_al[64], st_rs[64];

  const int tid = threadIdx.x, lane = tid & 31, w = tid >> 5;
  const int nl = lane & 15, hh = lane >> 4;
  const int hd = blockIdx.y, bb = blockIdx.z;
  const int qrow0 = blockIdx.x * 64 + w * 16;
  const size_t headoff = (size_t)hd * DD;

  // load Q fragments (q pre-scaled by 1/sqrt(D) in QKV epilogue)
  Frag qa[2];
  {
    const size_t qb = ((size_t)(bb * SS + qrow0 + nl)) * EE + headoff;
#pragma unroll
    for (int dh = 0; dh < 2; ++dh) {
      qa[dh].u[0] = *(const u32x4*)&qin[qb + dh * 32 + 16 * hh];
      qa[dh].u[1] = *(const u32x4*)&qin[qb + dh * 32 + 16 * hh + 8];
    }
  }

  v8f o0 = {}, o1 = {}, o2 = {}, o3 = {};
  float m_old = -1e30f, lsum = 0.f, mn_reg = 0.f, alpha = 0.f;

  const int sr = tid >> 2;            // staging row 0..31
  const int sd = (tid & 3) * 16;      // staging d offset

  for (int kc = 0; kc < SS / 32; ++kc) {
    const int kbase = kc * 32;
    __syncthreads();
    {  // stage K (row major) and V (transposed) cooperatively
      const size_t g = ((size_t)(bb * SS + kbase + sr)) * EE + headoff + sd;
      *(u32x4*)&Ks[sr * 64 + sd]     = *(const u32x4*)&kin[g];
      *(u32x4*)&Ks[sr * 64 + sd + 8] = *(const u32x4*)&kin[g + 8];
      union { u32x4 u[2]; bf16 h[16]; } t;
      t.u[0] = *(const u32x4*)&vin[g];
      t.u[1] = *(const u32x4*)&vin[g + 8];
#pragma unroll
      for (int i = 0; i < 16; ++i) Vst[(sd + i) * 32 + sr] = t.h[i];
    }
    __syncthreads();

    // scores: S[16 x 32] = Q(16x64) * K^T, two 16x16 tiles, 2 d-halves each
    v8f a0 = {}, a1 = {};
#pragma unroll
    for (int dh = 0; dh < 2; ++dh) {
      Frag b0, b1;
      const bf16* kp0 = &Ks[(0 * 16 + nl) * 64 + dh * 32 + 16 * hh];
      b0.u[0] = *(const u32x4*)kp0;  b0.u[1] = *(const u32x4*)(kp0 + 8);
      const bf16* kp1 = &Ks[(1 * 16 + nl) * 64 + dh * 32 + 16 * hh];
      b1.u[0] = *(const u32x4*)kp1;  b1.u[1] = *(const u32x4*)(kp1 + 8);
      a0 = wmma_bf16(qa[dh], b0, a0);
      a1 = wmma_bf16(qa[dh], b1, a1);
    }

    // per-row chunk max (reduce across 16-lane halves)
    float mx[8];
#pragma unroll
    for (int r = 0; r < 8; ++r) {
      float m0 = fmaxf(a0[r], a1[r]);
      m0 = fmaxf(m0, __shfl_xor(m0, 1, 32));
      m0 = fmaxf(m0, __shfl_xor(m0, 2, 32));
      m0 = fmaxf(m0, __shfl_xor(m0, 4, 32));
      m0 = fmaxf(m0, __shfl_xor(m0, 8, 32));
      mx[r] = m0;
    }
    if (nl == 0) {
#pragma unroll
      for (int r = 0; r < 8; ++r) st_mc[w * 16 + r + 8 * hh] = mx[r];
    }
    __syncthreads();
    if (lane < 16) {
      const float mc = st_mc[w * 16 + lane];
      mn_reg = fmaxf(m_old, mc);
      alpha = __expf(m_old - mn_reg);
      st_mn[w * 16 + lane] = mn_reg;
      st_al[w * 16 + lane] = alpha;
    }
    __syncthreads();

    float rsum[8];
#pragma unroll
    for (int r = 0; r < 8; ++r) {
      const int row = r + 8 * hh;
      const float mrow = st_mn[w * 16 + row];
      const float ar   = st_al[w * 16 + row];
      const float p0 = __expf(a0[r] - mrow);
      const float p1 = __expf(a1[r] - mrow);
      Ps[(w * 16 + row) * 32 + nl]      = (bf16)p0;
      Ps[(w * 16 + row) * 32 + 16 + nl] = (bf16)p1;
      o0[r] *= ar; o1[r] *= ar; o2[r] *= ar; o3[r] *= ar;
      float s_ = p0 + p1;
      s_ += __shfl_xor(s_, 1, 32); s_ += __shfl_xor(s_, 2, 32);
      s_ += __shfl_xor(s_, 4, 32); s_ += __shfl_xor(s_, 8, 32);
      rsum[r] = s_;
    }
    if (nl == 0) {
#pragma unroll
      for (int r = 0; r < 8; ++r) st_rs[w * 16 + r + 8 * hh] = rsum[r];
    }
    __syncthreads();
    if (lane < 16) { lsum = lsum * alpha + st_rs[w * 16 + lane]; m_old = mn_reg; }

    // O += P(16x32) * V(32x64)  -- 4 d-tiles
    Frag pa;
    const bf16* pp = &Ps[(w * 16 + nl) * 32];
    pa.u[0] = *(const u32x4*)(pp + 16 * hh);
    pa.u[1] = *(const u32x4*)(pp + 16 * hh + 8);
#pragma unroll
    for (int dt = 0; dt < 4; ++dt) {
      Frag bv;
      const bf16* vp = &Vst[(dt * 16 + nl) * 32 + 16 * hh];
      bv.u[0] = *(const u32x4*)vp;  bv.u[1] = *(const u32x4*)(vp + 8);
      if (dt == 0) o0 = wmma_bf16(pa, bv, o0);
      else if (dt == 1) o1 = wmma_bf16(pa, bv, o1);
      else if (dt == 2) o2 = wmma_bf16(pa, bv, o2);
      else o3 = wmma_bf16(pa, bv, o3);
    }
  }

  if (lane < 16) st_rs[w * 16 + lane] = 1.f / lsum;
  __syncthreads();
#pragma unroll
  for (int r = 0; r < 8; ++r) {
    const int row = qrow0 + r + 8 * hh;
    const float li = st_rs[w * 16 + r + 8 * hh];
    const size_t base = ((size_t)(bb * SS + row)) * EE + headoff + nl;
    ctx[base + 0]  = (bf16)(o0[r] * li);
    ctx[base + 16] = (bf16)(o1[r] * li);
    ctx[base + 32] = (bf16)(o2[r] * li);
    ctx[base + 48] = (bf16)(o3[r] * li);
  }
}

// ---------------------------------------------------------------------------
// Fused residual + LayerNorm (wave per row of 512). Optional bf16 copy.
// ---------------------------------------------------------------------------
__global__ __launch_bounds__(256) void ln_kernel(const float* __restrict__ xa,
                                                 const float* __restrict__ xd,
                                                 const float* __restrict__ g,
                                                 const float* __restrict__ bta,
                                                 float* __restrict__ outF,
                                                 bf16* outB) {
  const int lane = threadIdx.x & 31, w = threadIdx.x >> 5;
  const size_t row = (size_t)blockIdx.x * 8 + w;
  const float* pa = xa + row * 512;
  const float* pb = xd + row * 512;
  float v[16];
  float s = 0.f;
#pragma unroll
  for (int i = 0; i < 16; ++i) {
    v[i] = pa[lane * 16 + i] + pb[lane * 16 + i];
    s += v[i];
  }
  s += __shfl_xor(s, 16, 32); s += __shfl_xor(s, 8, 32);
  s += __shfl_xor(s, 4, 32);  s += __shfl_xor(s, 2, 32);
  s += __shfl_xor(s, 1, 32);
  const float mean = s * (1.f / 512.f);
  float q = 0.f;
#pragma unroll
  for (int i = 0; i < 16; ++i) { const float d = v[i] - mean; q += d * d; }
  q += __shfl_xor(q, 16, 32); q += __shfl_xor(q, 8, 32);
  q += __shfl_xor(q, 4, 32);  q += __shfl_xor(q, 2, 32);
  q += __shfl_xor(q, 1, 32);
  const float rinv = rsqrtf(q * (1.f / 512.f) + 1e-5f);
#pragma unroll
  for (int i = 0; i < 16; ++i) {
    const int c = lane * 16 + i;
    const float y = (v[i] - mean) * rinv * g[c] + bta[c];
    outF[row * 512 + c] = y;
    if (outB) outB[row * 512 + c] = (bf16)y;
  }
}

// ---------------------------------------------------------------------------
// Gating: wave per token. logits = x1 . gate_w[n] + gate_b; top-2 -> 0/1 mask
// (softmax is monotone, so top-k of logits == top-k of gating probs).
// ---------------------------------------------------------------------------
__global__ __launch_bounds__(256) void gate_kernel(const float* __restrict__ x1,
                                                   const float* __restrict__ gw,
                                                   const float* __restrict__ gb,
                                                   float* __restrict__ member) {
  const int lane = threadIdx.x & 31, w = threadIdx.x >> 5;
  const size_t tok = (size_t)blockIdx.x * 8 + w;
  const float* px = x1 + tok * 512;
  float xr[16];
#pragma unroll
  for (int i = 0; i < 16; ++i) xr[i] = px[lane * 16 + i];
  float lg[NEXPE];
#pragma unroll
  for (int n = 0; n < NEXPE; ++n) {
    const float* pw = gw + n * 512 + lane * 16;
    float s = 0.f;
#pragma unroll
    for (int i = 0; i < 16; ++i) s += xr[i] * pw[i];
    s += __shfl_xor(s, 16, 32); s += __shfl_xor(s, 8, 32);
    s += __shfl_xor(s, 4, 32);  s += __shfl_xor(s, 2, 32);
    s += __shfl_xor(s, 1, 32);
    lg[n] = s + gb[n];
  }
  int i1 = 0;
#pragma unroll
  for (int n = 1; n < NEXPE; ++n) if (lg[n] > lg[i1]) i1 = n;
  int i2 = (i1 == 0) ? 1 : 0;
#pragma unroll
  for (int n = 0; n < NEXPE; ++n)
    if (n != i1 && lg[n] > lg[i2]) i2 = n;
  if (lane < NEXPE)
    member[tok * NEXPE + lane] = (lane == i1 || lane == i2) ? 1.f : 0.f;
}

// ---------------------------------------------------------------------------
// Host launcher
// ---------------------------------------------------------------------------
extern "C" void kernel_launch(void* const* d_in, const int* in_sizes, int n_in,
                              void* d_out, int out_size, void* d_ws,
                              size_t ws_size, hipStream_t stream) {
  (void)in_sizes; (void)n_in; (void)out_size; (void)ws_size;
  const float* x      = (const float*)d_in[0];
  const float* in_w   = (const float*)d_in[1];
  const float* in_b   = (const float*)d_in[2];
  const float* out_w  = (const float*)d_in[3];
  const float* out_b  = (const float*)d_in[4];
  const float* gate_w = (const float*)d_in[5];
  const float* gate_b = (const float*)d_in[6];
  const float* exp_w  = (const float*)d_in[7];
  const float* exp_b  = (const float*)d_in[8];
  const float* ln1g   = (const float*)d_in[9];
  const float* ln1b   = (const float*)d_in[10];
  const float* ln2g   = (const float*)d_in[11];
  const float* ln2b   = (const float*)d_in[12];

  char* ws = (char*)d_ws;
  size_t off = 0;
  auto alloc = [&](size_t bytes) -> void* {
    void* p = ws + off;
    off = (off + bytes + 255) & ~(size_t)255;
    return p;
  };
  const size_t NTOK = (size_t)BS_ * EE;                 // 8,388,608
  bf16*  xb    = (bf16*)alloc(NTOK * 2);                // xb, later x1 bf16
  bf16*  inwb  = (bf16*)alloc((size_t)1536 * 512 * 2);
  bf16*  outwb = (bf16*)alloc((size_t)512 * 512 * 2);
  bf16*  expwb = (bf16*)alloc((size_t)NEXPE * 512 * 512 * 2);
  bf16*  qb    = (bf16*)alloc(NTOK * 2);                // q, later ctx (alias)
  bf16*  kb    = (bf16*)alloc(NTOK * 2);
  bf16*  vb    = (bf16*)alloc(NTOK * 2);
  float* tmpF  = (float*)alloc(NTOK * 4);               // attn_out, later moe
  float* x1    = (float*)alloc(NTOK * 4);
  float* memb  = (float*)alloc((size_t)BS_ * NEXPE * 4);

  // 1) precision conversion of activations + weights
  cvt_bf16_kernel<<<(int)(NTOK / 256), 256, 0, stream>>>(x, xb, (int)NTOK);
  cvt_bf16_kernel<<<(1536 * 512) / 256, 256, 0, stream>>>(in_w, inwb, 1536 * 512);
  cvt_bf16_kernel<<<(512 * 512) / 256, 256, 0, stream>>>(out_w, outwb, 512 * 512);
  cvt_bf16_kernel<<<(NEXPE * 512 * 512) / 256, 256, 0, stream>>>(
      exp_w, expwb, NEXPE * 512 * 512);

  // 2) QKV projection (q pre-scaled by 1/sqrt(D)); TDM-staged GEMM
  gemm_tdm_kernel<0, 1><<<dim3(BS_ / 64, 1536 / 128), 256, 0, stream>>>(
      xb, inwb, in_b, nullptr, qb, kb, vb, nullptr);

  // 3) flash attention; ctx overlays q
  attn_fa_kernel<<<dim3(SS / 64, HH_, BB), 128, 0, stream>>>(qb, kb, vb, qb);

  // 4) output projection -> fp32
  gemm_tdm_kernel<1, 1><<<dim3(BS_ / 64, 512 / 128), 256, 0, stream>>>(
      qb, outwb, out_b, tmpF, nullptr, nullptr, nullptr, nullptr);

  // 5) x1 = LN(x + attn_out); also bf16 copy for expert GEMMs
  ln_kernel<<<BS_ / 8, 256, 0, stream>>>(x, tmpF, ln1g, ln1b, x1, xb);

  // 6) gating top-2 membership mask
  gate_kernel<<<BS_ / 8, 256, 0, stream>>>(x1, gate_w, gate_b, memb);

  // 7) MoE: all 16 experts, masked accumulation (moe overlays attn_out)
  gemm_tdm_kernel<2, NEXPE><<<dim3(BS_ / 64, 512 / 128), 256, 0, stream>>>(
      xb, expwb, exp_b, tmpF, nullptr, nullptr, nullptr, memb);

  // 8) out = LN(x1 + moe)
  ln_kernel<<<BS_ / 8, 256, 0, stream>>>(x1, tmpF, ln2g, ln2b, (float*)d_out,
                                         nullptr);
}
